// Glm4MoeDecoderLayer_27582279975512
// MI455X (gfx1250) — matrix operand order repaired and checked by reading, not compiled, not run on hardware
//
#include <hip/hip_runtime.h>
#include <hip/hip_bf16.h>
#include <math.h>

#define T_    2048
#define DM_   2048
#define NH_   16
#define NKV_  2
#define DH_   128
#define NE_   16
#define IMOE_ 768
#define ISH_  768
#define QKVN_ 2560            // (16 + 2*2) * 128
#define SOFT_SCALE 0.08838834764831845f   // 1/sqrt(128)
#define EPS_  1e-5f

typedef __bf16 bf16;
typedef __attribute__((ext_vector_type(16))) __bf16 bf16x16;
typedef __attribute__((ext_vector_type(8)))  __bf16 bf16x8;
typedef __attribute__((ext_vector_type(8)))  float  f32x8;
typedef __attribute__((ext_vector_type(4)))  float  f32x4;

// ---------------------------------------------------------------------------
// CDNA5 async global->LDS copy (ASYNCcnt tracked, no VGPR round trip).
// vdst carries the workgroup-relative LDS byte offset (= low 32 bits of the
// generic address, per the ISA LDS aperture truncation rule).
// ---------------------------------------------------------------------------
__device__ __forceinline__ void async_g2l_b128(void* lds_dst, const void* gsrc) {
  asm volatile("global_load_async_to_lds_b128 %0, %1, off"
               :
               : "v"((unsigned)(uintptr_t)lds_dst), "v"(gsrc)
               : "memory");
}
__device__ __forceinline__ void wait_async0() {
  asm volatile("s_wait_asynccnt 0x0" ::: "memory");
}

// ---------------------------------------------------------------------------
// WMMA helpers: v_wmma_f32_16x16x32_bf16, D = A(16x32) * B(32x16) + C
// ---------------------------------------------------------------------------
__device__ __forceinline__ f32x8 wmma_bf16(bf16x16 a, bf16x16 b, f32x8 c) {
  return __builtin_amdgcn_wmma_f32_16x16x32_bf16(false, a, false, b,
                                                 (short)0, c, false, false);
}

// A-frag (16x32 bf16) from row-major tile, row stride `ld` elements.
__device__ __forceinline__ bf16x16 frag_a(const bf16* p0, int ld) {
  int lane = threadIdx.x & 31;
  const bf16* p = p0 + (size_t)(lane & 15) * ld + ((lane >> 4) << 3);
  bf16x8 lo = *(const bf16x8*)p;
  bf16x8 hi = *(const bf16x8*)(p + 16);
  bf16x16 r;
#pragma unroll
  for (int i = 0; i < 8; ++i) { r[i] = lo[i]; r[i + 8] = hi[i]; }
  return r;
}

// B-frag (32x16 bf16) from an N-major (transposed) tile BT[n][k].
__device__ __forceinline__ bf16x16 frag_b(const bf16* p0, int ld) {
  int lane = threadIdx.x & 31;
  const bf16* p = p0 + (size_t)(lane & 15) * ld + ((lane >> 4) << 3);
  bf16x8 lo = *(const bf16x8*)p;
  bf16x8 hi = *(const bf16x8*)(p + 16);
  bf16x16 r;
#pragma unroll
  for (int i = 0; i < 8; ++i) { r[i] = lo[i]; r[i + 8] = hi[i]; }
  return r;
}

// ---------------------------------------------------------------------------
// Fused residual add + RMSNorm.
// ---------------------------------------------------------------------------
__global__ void add_rmsnorm_kernel(const float* __restrict__ x,
                                   const float* __restrict__ res_in,
                                   const float* __restrict__ w,
                                   float* __restrict__ res_out,
                                   bf16* __restrict__ h16,
                                   float* __restrict__ h32) {
  int t = blockIdx.x, tid = threadIdx.x;
  __shared__ float red[256];
  float loc[8];
  float ss = 0.f;
#pragma unroll
  for (int i = 0; i < 8; ++i) {
    int c = tid + i * 256;
    float v = x[(size_t)t * DM_ + c] + res_in[(size_t)t * DM_ + c];
    loc[i] = v;
    res_out[(size_t)t * DM_ + c] = v;
    ss += v * v;
  }
  red[tid] = ss;
  __syncthreads();
  for (int s = 128; s > 0; s >>= 1) {
    if (tid < s) red[tid] += red[tid + s];
    __syncthreads();
  }
  float rs = rsqrtf(red[0] / (float)DM_ + EPS_);
#pragma unroll
  for (int i = 0; i < 8; ++i) {
    int c = tid + i * 256;
    float hv = loc[i] * rs * w[c];
    h16[(size_t)t * DM_ + c] = (bf16)hv;
    if (h32) h32[(size_t)t * DM_ + c] = hv;
  }
}

// ---------------------------------------------------------------------------
// Plain bf16 WMMA GEMM, double-buffered LDS, async A-tile copies.
// C(f32, MxN) = A(bf16, MxK) * B(f32->bf16, KxN) + bias.  Tile 64x128x32.
// ---------------------------------------------------------------------------
__global__ void __launch_bounds__(128, 1)
gemm_plain_kernel(const bf16* __restrict__ A,
                  const float* __restrict__ Bw,
                  const float* __restrict__ bias,
                  float* __restrict__ C,
                  int M, int N, int K) {
  __shared__ __align__(16) bf16 As[2][64 * 32];
  __shared__ __align__(16) bf16 Bs[2][128 * 32];  // transposed: Bs[n][k]
  int n0 = blockIdx.x * 128, m0 = blockIdx.y * 64;
  int tid = threadIdx.x, wave = tid >> 5, lane = tid & 31;
  f32x8 zero8 = {0.f, 0.f, 0.f, 0.f, 0.f, 0.f, 0.f, 0.f};
  f32x8 acc[8];
#pragma unroll
  for (int i = 0; i < 8; ++i) acc[i] = zero8;

  int ar = tid >> 1, ach = (tid & 1) * 16;
  const bf16* aptr = A + (size_t)(m0 + ar) * K + ach;
  f32x4 breg[8];

  auto stage_a = [&](int k0, int buf) {
    async_g2l_b128(&As[buf][ar * 32 + ach], aptr + k0);
    async_g2l_b128(&As[buf][ar * 32 + ach + 8], aptr + k0 + 8);
  };
  auto load_b = [&](int k0) {
#pragma unroll
    for (int i = 0; i < 8; ++i) {
      int idx = tid + i * 128;
      int kk = idx >> 5, nn = (idx & 31) << 2;
      breg[i] = *(const f32x4*)(Bw + (size_t)(k0 + kk) * N + n0 + nn);
    }
  };
  auto store_b = [&](int buf) {
#pragma unroll
    for (int i = 0; i < 8; ++i) {
      int idx = tid + i * 128;
      int kk = idx >> 5, nn = (idx & 31) << 2;
#pragma unroll
      for (int j = 0; j < 4; ++j) Bs[buf][(nn + j) * 32 + kk] = (bf16)breg[i][j];
    }
  };

  stage_a(0, 0);
  load_b(0);
  store_b(0);
  wait_async0();
  __syncthreads();

  int nk = K >> 5;
  for (int kt = 0; kt < nk; ++kt) {
    int buf = kt & 1;
    if (kt + 1 < nk) {               // issue next tile's traffic first
      stage_a((kt + 1) * 32, buf ^ 1);
      load_b((kt + 1) * 32);
    }
    bf16x16 af = frag_a(&As[buf][wave * 16 * 32], 32);
#pragma unroll
    for (int nb = 0; nb < 8; ++nb) {
      bf16x16 bfrg = frag_b(&Bs[buf][nb * 16 * 32], 32);
      acc[nb] = wmma_bf16(af, bfrg, acc[nb]);
    }
    if (kt + 1 < nk) {
      store_b(buf ^ 1);
      wait_async0();
    }
    __syncthreads();
  }

  int col = lane & 15, rb = (lane >> 4) << 3;
#pragma unroll
  for (int nb = 0; nb < 8; ++nb) {
    int nc = n0 + nb * 16 + col;
    float bv = bias ? bias[nc] : 0.f;
#pragma unroll
    for (int r = 0; r < 8; ++r) {
      int row = m0 + wave * 16 + rb + r;
      C[(size_t)row * N + nc] = acc[nb][r] + bv;
    }
  }
}

// ---------------------------------------------------------------------------
// Fused GLU GEMM: act = silu(A*Wg) * (A*Wu) * routing_weight -> bf16.
// Double-buffered, async A tiles.  gathered=1 -> expert e=blockIdx.z.
// ---------------------------------------------------------------------------
__global__ void __launch_bounds__(128, 1)
gemm_glu_kernel(const bf16* __restrict__ A,
                const float* __restrict__ Wgu_all,
                bf16* __restrict__ out_all,
                int K, int I, int gathered,
                const int* __restrict__ tok,
                const float* __restrict__ wgt,
                const int* __restrict__ cnt, int Mfull) {
  __shared__ __align__(16) bf16 As[2][64 * 32];
  __shared__ __align__(16) bf16 Bg[2][64 * 32];
  __shared__ __align__(16) bf16 Bu[2][64 * 32];
  int e = blockIdx.z;
  const float* Wgu = Wgu_all + (size_t)e * K * (2 * I);
  bf16* outp = out_all + (size_t)e * Mfull * I;
  const int* tokE = gathered ? (tok + (size_t)e * Mfull) : nullptr;
  const float* wgtE = gathered ? (wgt + (size_t)e * Mfull) : nullptr;
  int M = gathered ? cnt[e] : Mfull;
  int m0 = blockIdx.y * 64;
  if (m0 >= M) return;
  int n0 = blockIdx.x * 64;
  int tid = threadIdx.x, wave = tid >> 5, lane = tid & 31;
  f32x8 zero8 = {0.f, 0.f, 0.f, 0.f, 0.f, 0.f, 0.f, 0.f};
  f32x8 accg[4], accu[4];
#pragma unroll
  for (int i = 0; i < 4; ++i) { accg[i] = zero8; accu[i] = zero8; }

  int ar = tid >> 1, ach = (tid & 1) * 16;
  int slot = m0 + ar;
  int arow = (slot < M) ? (gathered ? tokE[slot] : slot) : 0;
  const bf16* aptr = A + (size_t)arow * K + ach;
  f32x4 breg[8];

  auto stage_a = [&](int k0, int buf) {
    async_g2l_b128(&As[buf][ar * 32 + ach], aptr + k0);
    async_g2l_b128(&As[buf][ar * 32 + ach + 8], aptr + k0 + 8);
  };
  auto load_b = [&](int k0) {
#pragma unroll
    for (int i = 0; i < 8; ++i) {
      int idx = tid + i * 128;          // 2 halves * 32 rows * 16 float4
      int half = idx >> 9, rem = idx & 511;
      int kk = rem >> 4, nn = (rem & 15) << 2;
      breg[i] = *(const f32x4*)(Wgu + (size_t)(k0 + kk) * (2 * I) +
                                half * I + n0 + nn);
    }
  };
  auto store_b = [&](int buf) {
#pragma unroll
    for (int i = 0; i < 8; ++i) {
      int idx = tid + i * 128;
      int half = idx >> 9, rem = idx & 511;
      int kk = rem >> 4, nn = (rem & 15) << 2;
      bf16* dst = half ? Bu[buf] : Bg[buf];
#pragma unroll
      for (int j = 0; j < 4; ++j) dst[(nn + j) * 32 + kk] = (bf16)breg[i][j];
    }
  };

  stage_a(0, 0);
  load_b(0);
  store_b(0);
  wait_async0();
  __syncthreads();

  int nk = K >> 5;
  for (int kt = 0; kt < nk; ++kt) {
    int buf = kt & 1;
    if (kt + 1 < nk) {
      stage_a((kt + 1) * 32, buf ^ 1);
      load_b((kt + 1) * 32);
    }
    bf16x16 af = frag_a(&As[buf][wave * 16 * 32], 32);
#pragma unroll
    for (int nb = 0; nb < 4; ++nb) {
      bf16x16 bg = frag_b(&Bg[buf][nb * 16 * 32], 32);
      accg[nb] = wmma_bf16(af, bg, accg[nb]);
      bf16x16 bu = frag_b(&Bu[buf][nb * 16 * 32], 32);
      accu[nb] = wmma_bf16(af, bu, accu[nb]);
    }
    if (kt + 1 < nk) {
      store_b(buf ^ 1);
      wait_async0();
    }
    __syncthreads();
  }

  int col = lane & 15, rb = (lane >> 4) << 3;
#pragma unroll
  for (int nb = 0; nb < 4; ++nb) {
#pragma unroll
    for (int r = 0; r < 8; ++r) {
      int s = m0 + wave * 16 + rb + r;
      if (s < M) {
        float g = accg[nb][r], u = accu[nb][r];
        float act = g / (1.f + __expf(-g)) * u;
        if (gathered) act *= wgtE[s];
        outp[(size_t)s * I + n0 + nb * 16 + col] = (bf16)act;
      }
    }
  }
}

// ---------------------------------------------------------------------------
// Expert down-projection: compact slot rows -> atomic f32 scatter into C.
// ---------------------------------------------------------------------------
__global__ void __launch_bounds__(128, 1)
gemm_down_kernel(const bf16* __restrict__ act_all,
                 const float* __restrict__ Wd_all,
                 float* __restrict__ C, int K, int N,
                 const int* __restrict__ tok,
                 const int* __restrict__ cnt,
                 int Mfull, float scale) {
  __shared__ __align__(16) bf16 As[2][64 * 32];
  __shared__ __align__(16) bf16 Bs[2][128 * 32];
  int e = blockIdx.z;
  int M = cnt[e];
  int m0 = blockIdx.y * 64;
  if (m0 >= M) return;
  const bf16* A = act_all + (size_t)e * Mfull * K;
  const float* Bw = Wd_all + (size_t)e * K * N;
  const int* tokE = tok + (size_t)e * Mfull;
  int n0 = blockIdx.x * 128;
  int tid = threadIdx.x, wave = tid >> 5, lane = tid & 31;
  f32x8 zero8 = {0.f, 0.f, 0.f, 0.f, 0.f, 0.f, 0.f, 0.f};
  f32x8 acc[8];
#pragma unroll
  for (int i = 0; i < 8; ++i) acc[i] = zero8;

  int ar = tid >> 1, ach = (tid & 1) * 16;
  const bf16* aptr = A + (size_t)(m0 + ar) * K + ach;
  f32x4 breg[8];

  auto stage_a = [&](int k0, int buf) {
    async_g2l_b128(&As[buf][ar * 32 + ach], aptr + k0);
    async_g2l_b128(&As[buf][ar * 32 + ach + 8], aptr + k0 + 8);
  };
  auto load_b = [&](int k0) {
#pragma unroll
    for (int i = 0; i < 8; ++i) {
      int idx = tid + i * 128;
      int kk = idx >> 5, nn = (idx & 31) << 2;
      breg[i] = *(const f32x4*)(Bw + (size_t)(k0 + kk) * N + n0 + nn);
    }
  };
  auto store_b = [&](int buf) {
#pragma unroll
    for (int i = 0; i < 8; ++i) {
      int idx = tid + i * 128;
      int kk = idx >> 5, nn = (idx & 31) << 2;
#pragma unroll
      for (int j = 0; j < 4; ++j) Bs[buf][(nn + j) * 32 + kk] = (bf16)breg[i][j];
    }
  };

  stage_a(0, 0);
  load_b(0);
  store_b(0);
  wait_async0();
  __syncthreads();

  int nk = K >> 5;
  for (int kt = 0; kt < nk; ++kt) {
    int buf = kt & 1;
    if (kt + 1 < nk) {
      stage_a((kt + 1) * 32, buf ^ 1);
      load_b((kt + 1) * 32);
    }
    bf16x16 af = frag_a(&As[buf][wave * 16 * 32], 32);
#pragma unroll
    for (int nb = 0; nb < 8; ++nb) {
      bf16x16 bfrg = frag_b(&Bs[buf][nb * 16 * 32], 32);
      acc[nb] = wmma_bf16(af, bfrg, acc[nb]);
    }
    if (kt + 1 < nk) {
      store_b(buf ^ 1);
      wait_async0();
    }
    __syncthreads();
  }

  int col = lane & 15, rb = (lane >> 4) << 3;
#pragma unroll
  for (int nb = 0; nb < 8; ++nb) {
#pragma unroll
    for (int r = 0; r < 8; ++r) {
      int s = m0 + wave * 16 + rb + r;
      if (s < M) {
        int t = tokE[s];
        atomicAdd(&C[(size_t)t * N + n0 + nb * 16 + col], acc[nb][r] * scale);
      }
    }
  }
}

// ---------------------------------------------------------------------------
// Per-token QK RMSNorm + RoPE + bf16 split into [head][T][128] layouts.
// ---------------------------------------------------------------------------
__device__ __forceinline__ void head_norm_rope(const float* __restrict__ x,
                                               const float* __restrict__ w,
                                               int p, bf16* __restrict__ out,
                                               int lane) {
  float v0 = x[lane], v1 = x[lane + 32], v2 = x[lane + 64], v3 = x[lane + 96];
  float ss = v0 * v0 + v1 * v1 + v2 * v2 + v3 * v3;
#pragma unroll
  for (int m = 16; m >= 1; m >>= 1) ss += __shfl_xor(ss, m, 32);
  float rs = rsqrtf(ss / (float)DH_ + EPS_);
  v0 *= rs * w[lane];
  v1 *= rs * w[lane + 32];
  v2 *= rs * w[lane + 64];
  v3 *= rs * w[lane + 96];
  float inv = __powf(1.0e6f, -(float)lane / 32.f);
  float fr = (float)p * inv;
  float s, c;
  __sincosf(fr, &s, &c);
  float r0 = v0 * c - v1 * s;
  float r1 = v1 * c + v0 * s;
  out[lane] = (bf16)r0;
  out[lane + 32] = (bf16)r1;
  out[lane + 64] = (bf16)v2;
  out[lane + 96] = (bf16)v3;
}

__global__ void qk_rope_kernel(const float* __restrict__ qkv,
                               const int* __restrict__ pos,
                               const float* __restrict__ qn,
                               const float* __restrict__ kn,
                               bf16* __restrict__ Qo, bf16* __restrict__ Ko,
                               bf16* __restrict__ Vo) {
  int t = blockIdx.x, tid = threadIdx.x, wave = tid >> 5, lane = tid & 31;
  const float* row = qkv + (size_t)t * QKVN_;
  int p = pos[t];
#pragma unroll
  for (int it = 0; it < 3; ++it) {
    int slot = it * 8 + wave;     // 0..23 ; 16 q heads, 2 k, 2 v
    if (slot < NH_) {
      head_norm_rope(row + slot * DH_, qn, p,
                     Qo + ((size_t)slot * T_ + t) * DH_, lane);
    } else if (slot < NH_ + NKV_) {
      int kh = slot - NH_;
      head_norm_rope(row + NH_ * DH_ + kh * DH_, kn, p,
                     Ko + ((size_t)kh * T_ + t) * DH_, lane);
    } else if (slot < NH_ + 2 * NKV_) {
      int vh = slot - NH_ - NKV_;
      const float* src = row + (NH_ + NKV_) * DH_ + vh * DH_;
      bf16* dst = Vo + ((size_t)vh * T_ + t) * DH_;
#pragma unroll
      for (int i = 0; i < 4; ++i) dst[lane + i * 32] = (bf16)src[lane + i * 32];
    }
  }
}

// ---------------------------------------------------------------------------
// FlashAttention (causal, GQA 16q/2kv), double-buffered K/V tiles; K tiles
// arrive via async global->LDS; V tiles via register transpose.
// ---------------------------------------------------------------------------
__global__ void __launch_bounds__(128, 1)
flash_attn_kernel(const bf16* __restrict__ Q,
                  const bf16* __restrict__ Kb,
                  const bf16* __restrict__ Vb,
                  bf16* __restrict__ O) {
  __shared__ __align__(16) bf16 Kt[2][32 * 128];    // row-major keys x d
  __shared__ __align__(16) bf16 Vt[2][128 * 32];    // transposed: d x keys
  __shared__ __align__(16) bf16 Pt[4 * 16 * 32];    // per-wave P tile
  int h = blockIdx.y, qt = blockIdx.x;
  int kvh = h >> 3;
  int tid = threadIdx.x, wave = tid >> 5, lane = tid & 31;
  int qbase = qt * 64 + wave * 16;
  int col = lane & 15, rb = (lane >> 4) << 3;

  bf16x16 qf[4];
  {
    int row = lane & 15, kb8 = (lane >> 4) << 3;
    const bf16* qp = Q + ((size_t)h * T_ + qbase + row) * DH_ + kb8;
#pragma unroll
    for (int c = 0; c < 4; ++c) {
      bf16x8 lo = *(const bf16x8*)(qp + c * 32);
      bf16x8 hi = *(const bf16x8*)(qp + c * 32 + 16);
#pragma unroll
      for (int i = 0; i < 8; ++i) { qf[c][i] = lo[i]; qf[c][8 + i] = hi[i]; }
    }
  }
  f32x8 zero8 = {0.f, 0.f, 0.f, 0.f, 0.f, 0.f, 0.f, 0.f};
  f32x8 oacc[8];
#pragma unroll
  for (int i = 0; i < 8; ++i) oacc[i] = zero8;
  float mrow[8], lrow[8];
#pragma unroll
  for (int r = 0; r < 8; ++r) { mrow[r] = -1e30f; lrow[r] = 0.f; }

  int vj = tid & 31, vd0 = (tid >> 5) * 32;
  bf16x8 vreg[4];
  auto stage_k = [&](int k0, int buf) {
#pragma unroll
    for (int i = 0; i < 4; ++i) {
      int idx = tid + i * 128;           // 512 chunks of 8 bf16
      int j = idx >> 4, dc = (idx & 15) << 3;
      async_g2l_b128(&Kt[buf][j * 128 + dc],
                     Kb + ((size_t)kvh * T_ + k0 + j) * DH_ + dc);
    }
  };
  auto load_v = [&](int k0) {
    const bf16* vp = Vb + ((size_t)kvh * T_ + k0 + vj) * DH_ + vd0;
#pragma unroll
    for (int i = 0; i < 4; ++i) vreg[i] = *(const bf16x8*)(vp + i * 8);
  };
  auto store_v = [&](int buf) {
#pragma unroll
    for (int i = 0; i < 4; ++i)
#pragma unroll
      for (int q = 0; q < 8; ++q)
        Vt[buf][(vd0 + i * 8 + q) * 32 + vj] = vreg[i][q];
  };

  int ntiles = qt * 2 + 2;
  stage_k(0, 0);
  load_v(0);
  store_v(0);
  wait_async0();
  __syncthreads();

  for (int jt = 0; jt < ntiles; ++jt) {
    int buf = jt & 1;
    int k0 = jt * 32;
    if (jt + 1 < ntiles) {
      stage_k((jt + 1) * 32, buf ^ 1);
      load_v((jt + 1) * 32);
    }
    // S = Q K^T
    f32x8 s0 = zero8, s1 = zero8;
#pragma unroll
    for (int c = 0; c < 4; ++c) {
      bf16x16 kf0 = frag_b(&Kt[buf][0 * 16 * 128 + c * 32], 128);
      s0 = wmma_bf16(qf[c], kf0, s0);
      bf16x16 kf1 = frag_b(&Kt[buf][1 * 16 * 128 + c * 32], 128);
      s1 = wmma_bf16(qf[c], kf1, s1);
    }
    // online softmax
#pragma unroll
    for (int r = 0; r < 8; ++r) {
      int t = qbase + rb + r;
      float a = s0[r] * SOFT_SCALE;
      float b = s1[r] * SOFT_SCALE;
      a = (k0 + col) <= t ? a : -1e30f;
      b = (k0 + 16 + col) <= t ? b : -1e30f;
      float mr = fmaxf(a, b);
#pragma unroll
      for (int m = 8; m >= 1; m >>= 1) mr = fmaxf(mr, __shfl_xor(mr, m, 32));
      float mnew = fmaxf(mrow[r], mr);
      float p0 = __expf(a - mnew), p1 = __expf(b - mnew);
      float rsum = p0 + p1;
#pragma unroll
      for (int m = 8; m >= 1; m >>= 1) rsum += __shfl_xor(rsum, m, 32);
      float corr = __expf(mrow[r] - mnew);
      lrow[r] = lrow[r] * corr + rsum;
      mrow[r] = mnew;
#pragma unroll
      for (int nb = 0; nb < 8; ++nb) oacc[nb][r] *= corr;
      bf16* pp = Pt + wave * 16 * 32 + (rb + r) * 32;
      pp[col] = (bf16)p0;
      pp[16 + col] = (bf16)p1;
    }
    // wave-private LDS round trip (ds ops in-order within a wave)
    bf16x16 pf = frag_a(Pt + wave * 16 * 32, 32);
#pragma unroll
    for (int nb = 0; nb < 8; ++nb) {
      bf16x16 vf = frag_b(&Vt[buf][nb * 16 * 32], 32);
      oacc[nb] = wmma_bf16(pf, vf, oacc[nb]);
    }
    if (jt + 1 < ntiles) {
      store_v(buf ^ 1);
      wait_async0();
    }
    __syncthreads();
  }
#pragma unroll
  for (int r = 0; r < 8; ++r) {
    float inv = 1.f / lrow[r];
    int t = qbase + rb + r;
    bf16* op = O + (size_t)t * (NH_ * DH_) + h * DH_;
#pragma unroll
    for (int nb = 0; nb < 8; ++nb)
      op[nb * 16 + col] = (bf16)(oacc[nb][r] * inv);
  }
}

// ---------------------------------------------------------------------------
// Router: sigmoid gate, group top-2-sum, top-2 groups, top-4 experts,
// normalized weights; builds per-expert token lists with atomics.
// ---------------------------------------------------------------------------
__global__ void route_kernel(const float* __restrict__ h2,
                             const float* __restrict__ gw,
                             const float* __restrict__ gbias,
                             int* __restrict__ cnt, int* __restrict__ tok,
                             float* __restrict__ wgt) {
  int t = blockIdx.x, tid = threadIdx.x;
  __shared__ float part[256 * NE_];
  float loc[NE_];
#pragma unroll
  for (int e = 0; e < NE_; ++e) loc[e] = 0.f;
  for (int i = 0; i < 8; ++i) {
    int c = tid + i * 256;
    float x = h2[(size_t)t * DM_ + c];
    const float* g = gw + (size_t)c * NE_;
#pragma unroll
    for (int e = 0; e < NE_; ++e) loc[e] += x * g[e];
  }
#pragma unroll
  for (int e = 0; e < NE_; ++e) part[tid * NE_ + e] = loc[e];
  __syncthreads();
  if (tid < NE_) {
    float s = 0.f;
    for (int j = 0; j < 256; ++j) s += part[j * NE_ + tid];
    part[tid] = s;
  }
  __syncthreads();
  if (tid == 0) {
    float sc[NE_], sb[NE_];
#pragma unroll
    for (int e = 0; e < NE_; ++e) {
      sc[e] = 1.f / (1.f + __expf(-part[e]));
      sb[e] = sc[e] + gbias[e];
    }
    float gsc[4];
#pragma unroll
    for (int g = 0; g < 4; ++g) {
      float m1 = -1e30f, m2 = -1e30f;
#pragma unroll
      for (int i = 0; i < 4; ++i) {
        float v = sb[4 * g + i];
        if (v > m1) { m2 = m1; m1 = v; }
        else if (v > m2) m2 = v;
      }
      gsc[g] = m1 + m2;
    }
    int g1 = 0;
    for (int g = 1; g < 4; ++g) if (gsc[g] > gsc[g1]) g1 = g;
    int g2 = -1;
    for (int g = 0; g < 4; ++g)
      if (g != g1 && (g2 < 0 || gsc[g] > gsc[g2])) g2 = g;
    float ms[NE_];
#pragma unroll
    for (int e = 0; e < NE_; ++e) {
      int g = e >> 2;
      ms[e] = (g == g1 || g == g2) ? sb[e] : -1e30f;
    }
    int idx[4];
    float wv[4];
    float wsum = 0.f;
    bool used[NE_];
#pragma unroll
    for (int e = 0; e < NE_; ++e) used[e] = false;
    for (int k = 0; k < 4; ++k) {
      int best = -1;
      for (int e = 0; e < NE_; ++e)
        if (!used[e] && (best < 0 || ms[e] > ms[best])) best = e;
      used[best] = true;
      idx[k] = best;
      wv[k] = sc[best];
      wsum += wv[k];
    }
    for (int k = 0; k < 4; ++k) {
      float ww = wv[k] / wsum;
      int s = atomicAdd(&cnt[idx[k]], 1);
      tok[(size_t)idx[k] * T_ + s] = t;
      wgt[(size_t)idx[k] * T_ + s] = ww;
    }
  }
}

// ---------------------------------------------------------------------------
// Host-side orchestration
// ---------------------------------------------------------------------------
extern "C" void kernel_launch(void* const* d_in, const int* in_sizes, int n_in,
                              void* d_out, int out_size, void* d_ws,
                              size_t ws_size, hipStream_t stream) {
  const int* positions    = (const int*)d_in[0];
  const float* hidden     = (const float*)d_in[1];
  const float* residual   = (const float*)d_in[2];
  const float* ln1_w      = (const float*)d_in[3];
  const float* ln2_w      = (const float*)d_in[4];
  const float* w_qkv      = (const float*)d_in[5];
  const float* b_qkv      = (const float*)d_in[6];
  const float* w_o        = (const float*)d_in[7];
  const float* q_norm_w   = (const float*)d_in[8];
  const float* k_norm_w   = (const float*)d_in[9];
  const float* gate_w     = (const float*)d_in[10];
  const float* gate_bias  = (const float*)d_in[11];
  const float* shared_wgu = (const float*)d_in[12];
  const float* shared_wd  = (const float*)d_in[13];
  const float* expert_wgu = (const float*)d_in[14];
  const float* expert_wd  = (const float*)d_in[15];

  float* h_out = (float*)d_out;                         // [T, DM]
  float* resid_out = (float*)d_out + (size_t)T_ * DM_;  // [T, DM]

  char* base = (char*)d_ws;
  size_t off = 0;
  auto take = [&](size_t bytes) -> char* {
    char* p = base + off;
    off = (off + bytes + 255) & ~(size_t)255;
    return p;
  };
  float* resid1  = (float*)take((size_t)T_ * DM_ * 4);
  bf16*  h1      = (bf16*)take((size_t)T_ * DM_ * 2);
  float* qkv     = (float*)take((size_t)T_ * QKVN_ * 4);
  bf16*  qb      = (bf16*)take((size_t)NH_ * T_ * DH_ * 2);
  bf16*  kb      = (bf16*)take((size_t)NKV_ * T_ * DH_ * 2);
  bf16*  vb      = (bf16*)take((size_t)NKV_ * T_ * DH_ * 2);
  bf16*  ob      = (bf16*)take((size_t)T_ * NH_ * DH_ * 2);
  float* attn    = (float*)take((size_t)T_ * DM_ * 4);
  float* h2f     = (float*)take((size_t)T_ * DM_ * 4);
  bf16*  h2b     = (bf16*)take((size_t)T_ * DM_ * 2);
  bf16*  act_sh  = (bf16*)take((size_t)T_ * ISH_ * 2);
  bf16*  act_ex  = (bf16*)take((size_t)NE_ * T_ * IMOE_ * 2);
  int*   cnt     = (int*)take((size_t)NE_ * 4);
  int*   tok     = (int*)take((size_t)NE_ * T_ * 4);
  float* wgt     = (float*)take((size_t)NE_ * T_ * 4);
  (void)ws_size; (void)in_sizes; (void)n_in; (void)out_size;

  add_rmsnorm_kernel<<<T_, 256, 0, stream>>>(hidden, residual, ln1_w, resid1,
                                             h1, nullptr);
  gemm_plain_kernel<<<dim3(QKVN_ / 128, T_ / 64), 128, 0, stream>>>(
      h1, w_qkv, b_qkv, qkv, T_, QKVN_, DM_);
  qk_rope_kernel<<<T_, 256, 0, stream>>>(qkv, positions, q_norm_w, k_norm_w,
                                         qb, kb, vb);
  flash_attn_kernel<<<dim3(T_ / 64, NH_), 128, 0, stream>>>(qb, kb, vb, ob);
  gemm_plain_kernel<<<dim3(DM_ / 128, T_ / 64), 128, 0, stream>>>(
      ob, w_o, nullptr, attn, T_, DM_, NH_ * DH_);
  add_rmsnorm_kernel<<<T_, 256, 0, stream>>>(attn, resid1, ln2_w, resid_out,
                                             h2b, h2f);
  hipMemsetAsync(cnt, 0, NE_ * sizeof(int), stream);
  route_kernel<<<T_, 256, 0, stream>>>(h2f, gate_w, gate_bias, cnt, tok, wgt);
  gemm_glu_kernel<<<dim3(ISH_ / 64, T_ / 64, 1), 128, 0, stream>>>(
      h2b, shared_wgu, act_sh, DM_, ISH_, 0, nullptr, nullptr, nullptr, T_);
  gemm_plain_kernel<<<dim3(DM_ / 128, T_ / 64), 128, 0, stream>>>(
      act_sh, shared_wd, nullptr, h_out, T_, DM_, ISH_);
  gemm_glu_kernel<<<dim3(IMOE_ / 64, T_ / 64, NE_), 128, 0, stream>>>(
      h2b, expert_wgu, act_ex, DM_, IMOE_, 1, tok, wgt, cnt, T_);
  gemm_down_kernel<<<dim3(DM_ / 128, T_ / 64, NE_), 128, 0, stream>>>(
      act_ex, expert_wd, h_out, IMOE_, DM_, tok, cnt, T_, 1.0f /*ROUTED_SCALE*/);
}
